// ScalingModel_35270271435267
// MI455X (gfx1250) — matrix operand order repaired and checked by reading, not compiled, not run on hardware
//
#include <hip/hip_runtime.h>
#include <hip/hip_bf16.h>
#include <stdint.h>

#define Hn    256
#define H2n   512
#define Bn    32
#define Tn    4096
#define Vn    32000
#define Mslots 512
#define FSn   384
#define NCAND (Tn - 3)        // 4093
#define RETRO (Mslots - FSn)  // 128
#define EPSF  1e-5f

typedef __attribute__((ext_vector_type(16))) __bf16 v16bf;
typedef __attribute__((ext_vector_type(8)))  float  v8f;

static __device__ __forceinline__ __bf16 tobf(float f) { return (__bf16)f; }

static __device__ __forceinline__ float wred32(float v) {
#pragma unroll
  for (int m = 16; m >= 1; m >>= 1) v += __shfl_xor(v, m, 32);
  return v;
}
static __device__ __forceinline__ float wred16(float v) {
#pragma unroll
  for (int m = 8; m >= 1; m >>= 1) v += __shfl_xor(v, m, 32);
  return v;
}

static __device__ __forceinline__ v8f wmma_bf16(v16bf a, v16bf b, v8f c) {
  // D = A(16x32 bf16) * B(32x16 bf16) + C(16x16 f32)
  return __builtin_amdgcn_wmma_f32_16x16x32_bf16(false, a, false, b, (short)0, c,
                                                 false, false);
}

// A-fragment (16x32, bf16) from an f32 LDS tile. ISA 7.12.2 16-bit A layout:
// lanes 0-15: M=lane, V0..3 K=kh*8+2v, V4..7 K=16+kh*8+2(v-4); lanes 16-31 kh=1.
static __device__ __forceinline__ v16bf afrag_f32(const float* tile, int stride,
                                                  int kc, int lane) {
  int m = lane & 15, kh = lane >> 4;
  const float* row = tile + m * stride + kc;
  v16bf a;
#pragma unroll
  for (int v = 0; v < 8; ++v) {
    int kv = (v < 4) ? (kh * 8 + 2 * v) : (16 + kh * 8 + 2 * (v - 4));
    a[2 * v]     = tobf(row[kv]);
    a[2 * v + 1] = tobf(row[kv + 1]);
  }
  return a;
}
static __device__ __forceinline__ v16bf afrag_bf16(const __bf16* tile, int stride,
                                                   int kc, int lane) {
  int m = lane & 15, kh = lane >> 4;
  const __bf16* row = tile + m * stride + kc;
  v16bf a;
#pragma unroll
  for (int v = 0; v < 8; ++v) {
    int kv = (v < 4) ? (kh * 8 + 2 * v) : (16 + kh * 8 + 2 * (v - 4));
    a[2 * v]     = row[kv];
    a[2 * v + 1] = row[kv + 1];
  }
  return a;
}
// Packed-B fragment: one 32-byte vector load per lane (2x global_load_b128).
static __device__ __forceinline__ v16bf bfrag_packed(const __bf16* Wp, int tile,
                                                     int lane) {
  return *(const v16bf*)(Wp + ((size_t)tile * 32 + lane) * 16);
}

// -------- pack f32 weights [K][N] into WMMA B-fragment tile order (bf16) -----
// tile = nt*(K/32) + kt ; within tile: lane-major, 16 contiguous bf16 per lane.
__global__ __launch_bounds__(256) void pack_b(const float* W, __bf16* out,
                                              int Kdim, int Ndim) {
  int gid = blockIdx.x * blockDim.x + threadIdx.x;
  int KT = Kdim >> 5;
  int total = KT * (Ndim >> 4) * 32;
  if (gid >= total) return;
  int lane = gid & 31;
  int tile = gid >> 5;
  int kt = tile % KT, nt = tile / KT;
  int n0 = nt * 16 + (lane & 15);
  int kh = lane >> 4;
  v16bf f;
#pragma unroll
  for (int v = 0; v < 8; ++v) {
    int k = kt * 32 + ((v < 4) ? (kh * 8 + 2 * v) : (16 + kh * 8 + 2 * (v - 4)));
    f[2 * v]     = tobf(W[k * Ndim + n0]);
    f[2 * v + 1] = tobf(W[(k + 1) * Ndim + n0]);
  }
  *(v16bf*)(out + (size_t)gid * 16) = f;
}

// -------------------------------------------- fused embed+FFN+LN+score+context
__global__ __launch_bounds__(128) void ffn_ln_kernel(
    const int* seq, const float* embed, const float* b1, const float* b2,
    const float* ln_g, const float* ln_b, const float* Wg, const float* bg,
    const __bf16* W1p, const __bf16* W2p, float* hidden, float* fscore,
    float* ctxsum) {
  extern __shared__ char smem[];
  const int lane = threadIdx.x & 31, wid = threadIdx.x >> 5;
  float*  hT   = (float*)(smem + wid * (16 * Hn * 4 + 16 * H2n * 2));
  __bf16* act1 = (__bf16*)(hT + 16 * Hn);
  const int tile = blockIdx.x * (blockDim.x >> 5) + wid;
  const int b = tile >> 8;              // 256 tiles per batch (T/16)
  const int tok0 = (tile & 255) * 16;
  // 1) gather 16 embedding rows -> LDS (f32)
  for (int it = 0; it < 32; ++it) {
    int idx = (it * 32 + lane) * 4;
    int r = idx >> 8, c = idx & 255;
    int row = seq[b * Tn + tok0 + r];
    *(float4*)(hT + r * Hn + c) = *(const float4*)(embed + (long)row * Hn + c);
  }
  __syncthreads();
  const int nlo = lane & 15, kh = lane >> 4;
  // 2) GEMM1 (16xH)x(HxH2) -> relu -> act1 (bf16); dual accumulator chains
  v16bf a1[8];
#pragma unroll
  for (int kt = 0; kt < 8; ++kt) a1[kt] = afrag_f32(hT, Hn, kt * 32, lane);
  for (int nt = 0; nt < 32; ++nt) {
    int n0 = nt * 16 + nlo;
    float bias = b1[n0];
    v8f acc0, acc1;
#pragma unroll
    for (int v = 0; v < 8; ++v) { acc0[v] = bias; acc1[v] = 0.f; }
#pragma unroll
    for (int kt = 0; kt < 8; kt += 2) {
      acc0 = wmma_bf16(a1[kt],     bfrag_packed(W1p, nt * 8 + kt,     lane), acc0);
      acc1 = wmma_bf16(a1[kt + 1], bfrag_packed(W1p, nt * 8 + kt + 1, lane), acc1);
    }
#pragma unroll
    for (int v = 0; v < 8; ++v) {
      float s = acc0[v] + acc1[v];
      act1[(v + 8 * kh) * H2n + n0] = tobf(s > 0.f ? s : 0.f);
    }
  }
  __syncthreads();
  // 3) GEMM2 (16xH2)x(H2xH) + residual, x written in place into hT
  v16bf a2[16];
#pragma unroll
  for (int kt = 0; kt < 16; ++kt) a2[kt] = afrag_bf16(act1, H2n, kt * 32, lane);
  for (int nt = 0; nt < 16; ++nt) {
    int n0 = nt * 16 + nlo;
    float bias = b2[n0];
    v8f acc0, acc1;
#pragma unroll
    for (int v = 0; v < 8; ++v) { acc0[v] = bias; acc1[v] = 0.f; }
#pragma unroll
    for (int kt = 0; kt < 16; kt += 2) {
      acc0 = wmma_bf16(a2[kt],     bfrag_packed(W2p, nt * 16 + kt,     lane), acc0);
      acc1 = wmma_bf16(a2[kt + 1], bfrag_packed(W2p, nt * 16 + kt + 1, lane), acc1);
    }
#pragma unroll
    for (int v = 0; v < 8; ++v) hT[(v + 8 * kh) * Hn + n0] += acc0[v] + acc1[v];
  }
  __syncthreads();
  // 4) LayerNorm + fwd score + context partial sums
  float lg[8], lb[8], wg[8], ctxp[8];
#pragma unroll
  for (int j = 0; j < 8; ++j) {
    int c = lane + 32 * j;
    lg[j] = ln_g[c]; lb[j] = ln_b[c]; wg[j] = Wg[c]; ctxp[j] = 0.f;
  }
  float bgv = bg[0];
  for (int m = 0; m < 16; ++m) {
    float xv[8], s1 = 0.f, s2 = 0.f;
#pragma unroll
    for (int j = 0; j < 8; ++j) {
      float x = hT[m * Hn + lane + 32 * j];
      xv[j] = x; s1 += x; s2 += x * x;
    }
    s1 = wred32(s1); s2 = wred32(s2);
    float mu = s1 * (1.f / Hn);
    float rstd = rsqrtf(s2 * (1.f / Hn) - mu * mu + EPSF);
    float sc = 0.f;
    long gbase = (long)(b * Tn + tok0 + m) * Hn;
#pragma unroll
    for (int j = 0; j < 8; ++j) {
      float val = (xv[j] - mu) * rstd * lg[j] + lb[j];
      hidden[gbase + lane + 32 * j] = val;
      sc += val * wg[j];
      ctxp[j] += val;
    }
    sc = wred32(sc);
    if (lane == 0) fscore[b * Tn + tok0 + m] = sc + bgv;
  }
#pragma unroll
  for (int j = 0; j < 8; ++j) atomicAdd(&ctxsum[b * Hn + lane + 32 * j], ctxp[j]);
}

__global__ __launch_bounds__(256) void finalize_ctx(const float* ctxsum, float* context) {
  int i = blockIdx.x * blockDim.x + threadIdx.x;
  if (i < Bn * Hn) context[i] = ctxsum[i] * (1.f / Tn);
}

// ------------------------------------------------ gate MLP -> retro scores
__global__ __launch_bounds__(128) void gate_kernel(
    const float* hidden, const float* context, const __bf16* Wr1p,
    const float* br1, const float* Wr2, const float* br2, float* rscores) {
  extern __shared__ char smem[];
  const int lane = threadIdx.x & 31, wid = threadIdx.x >> 5;
  float* gT = (float*)(smem + wid * (16 * Hn * 4 + Hn * 4));
  float* cT = gT + 16 * Hn;
  const int tile = blockIdx.x * (blockDim.x >> 5) + wid;
  const int b = tile >> 8;
  const int tok0 = (tile & 255) * 16;
  for (int it = 0; it < 32; ++it) {
    int idx = (it * 32 + lane) * 4;
    int r = idx >> 8, c = idx & 255;
    int tok = tok0 + r; if (tok > NCAND - 1) tok = NCAND - 1;
    *(float4*)(gT + r * Hn + c) = *(const float4*)(hidden + (long)(b * Tn + tok) * Hn + c);
  }
#pragma unroll
  for (int j = 0; j < 2; ++j) {
    int c = (j * 32 + lane) * 4;
    *(float4*)(cT + c) = *(const float4*)(context + b * Hn + c);
  }
  __syncthreads();
  const int nlo = lane & 15, kh = lane >> 4;
  v16bf a[16];
#pragma unroll
  for (int kt = 0; kt < 8; ++kt) a[kt] = afrag_f32(gT, Hn, kt * 32, lane);
#pragma unroll
  for (int kt = 0; kt < 8; ++kt) {   // K 256..511: broadcast context rows
    const float* rowc = cT + kt * 32;
    v16bf f;
#pragma unroll
    for (int v = 0; v < 8; ++v) {
      int kv = (v < 4) ? (kh * 8 + 2 * v) : (16 + kh * 8 + 2 * (v - 4));
      f[2 * v] = tobf(rowc[kv]); f[2 * v + 1] = tobf(rowc[kv + 1]);
    }
    a[8 + kt] = f;
  }
  float rowacc[8] = {0, 0, 0, 0, 0, 0, 0, 0};
  for (int nt = 0; nt < 16; ++nt) {
    int n0 = nt * 16 + nlo;
    float bias = br1[n0];
    v8f acc0, acc1;
#pragma unroll
    for (int v = 0; v < 8; ++v) { acc0[v] = bias; acc1[v] = 0.f; }
#pragma unroll
    for (int kt = 0; kt < 16; kt += 2) {
      acc0 = wmma_bf16(a[kt],     bfrag_packed(Wr1p, nt * 16 + kt,     lane), acc0);
      acc1 = wmma_bf16(a[kt + 1], bfrag_packed(Wr1p, nt * 16 + kt + 1, lane), acc1);
    }
    float w2 = Wr2[n0];
#pragma unroll
    for (int v = 0; v < 8; ++v) {
      float r = acc0[v] + acc1[v];
      rowacc[v] += (r > 0.f ? r : 0.f) * w2;   // fold dot(Wr2) into epilogue
    }
  }
#pragma unroll
  for (int v = 0; v < 8; ++v) rowacc[v] = wred16(rowacc[v]);
  float br2v = br2[0];
  if (nlo < 8) {
    int m = kh * 8 + nlo;
    int tok = tok0 + m;
    if (tok < NCAND) {
      float sv = rowacc[0];
#pragma unroll
      for (int v = 1; v < 8; ++v) if (nlo == v) sv = rowacc[v];
      rscores[b * Tn + tok] = 1.f / (1.f + expf(-(sv + br2v)));
    }
  }
}

// ------------------------------------------ exact top-k via threshold search
__global__ __launch_bounds__(256) void topk_kernel(
    const float* scores, const unsigned char* selin, unsigned char* selout,
    int* slot_idx, int slotoff, int n, int k) {
  __shared__ unsigned int cnt;
  __shared__ int gcnt[256], ecnt[256];
  const int b = blockIdx.x, t = threadIdx.x;
  const float* s = scores + b * Tn;
  const unsigned char* si = selin ? (selin + b * Tn) : nullptr;
  auto key = [&](int i) -> unsigned int {
    if (si && si[i]) return 0u;    // excluded (== -inf)
    unsigned int u = __float_as_uint(s[i]);
    return (u & 0x80000000u) ? ~u : (u | 0x80000000u);
  };
  unsigned long long lo = 0ull, hi = 0xFFFFFFFFull;
  while (lo < hi) {
    unsigned int mid = (unsigned int)((lo + hi + 1ull) >> 1);
    if (t == 0) cnt = 0;
    __syncthreads();
    unsigned int lc = 0;
    for (int i = t; i < n; i += 256) lc += (key(i) >= mid) ? 1u : 0u;
    atomicAdd(&cnt, lc);
    __syncthreads();
    unsigned int c = cnt;
    __syncthreads();
    if (c >= (unsigned)k) lo = mid; else hi = mid - 1;
  }
  unsigned int thr = (unsigned int)lo;
  if (t == 0) cnt = 0;
  __syncthreads();
  { unsigned int lc = 0;
    for (int i = t; i < n; i += 256) lc += (key(i) > thr) ? 1u : 0u;
    atomicAdd(&cnt, lc); }
  __syncthreads();
  int need = k - (int)cnt;        // ties taken in ascending index order
  __syncthreads();
  const int chunk = (n + 255) >> 8;
  int beg = t * chunk, end = beg + chunk; if (end > n) end = n;
  int g = 0, e = 0;
  for (int i = beg; i < end; ++i) { unsigned int u = key(i); g += (u > thr); e += (u == thr); }
  gcnt[t] = g; ecnt[t] = e;
  __syncthreads();
  if (t == 0) {
    int ga = 0, ea = 0;
    for (int i = 0; i < 256; ++i) {
      int tg = gcnt[i], te = ecnt[i];
      gcnt[i] = ga; ecnt[i] = ea; ga += tg; ea += te;
    }
  }
  __syncthreads();
  int gr = gcnt[t], er = ecnt[t];
  int* out = slot_idx + b * Mslots + slotoff;
  for (int i = beg; i < end; ++i) {
    unsigned int u = key(i);
    if (u > thr) {
      int pos = gr + (er < need ? er : need);
      out[pos] = i;
      if (selout) selout[b * Tn + i] = 1;
      ++gr;
    } else if (u == thr) {
      if (er < need) {
        out[gr + er] = i;
        if (selout) selout[b * Tn + i] = 1;
      }
      ++er;
    }
  }
}

// --------------------------------------------- q-proj + attention over slots
__global__ __launch_bounds__(256) void attn_kernel(
    const float* hidden, const int* slot_idx, const float* Wq, const float* bq,
    float* ctxv) {
  __shared__ float hrow[Hn], qv[Hn], sc[Mslots], red[256];
  __shared__ int sidx[Mslots];
  const int b = blockIdx.x, t = threadIdx.x;
  hrow[t] = hidden[(long)(b * Tn + Tn - 2) * Hn + t];
  sidx[t] = slot_idx[b * Mslots + t];
  sidx[t + 256] = slot_idx[b * Mslots + t + 256];
  __syncthreads();
  float q = bq[t];
  for (int k2 = 0; k2 < Hn; ++k2) q += hrow[k2] * Wq[k2 * Hn + t];
  qv[t] = q;
  __syncthreads();
  float smax = -1e30f;
  for (int m = t; m < Mslots; m += 256) {
    const float* row = hidden + (long)(b * Tn + sidx[m]) * Hn;
    float d = 0.f;
    for (int k2 = 0; k2 < Hn; ++k2) d += row[k2] * qv[k2];
    sc[m] = d;
    smax = fmaxf(smax, d);
  }
  red[t] = smax; __syncthreads();
  for (int st = 128; st >= 1; st >>= 1) { if (t < st) red[t] = fmaxf(red[t], red[t + st]); __syncthreads(); }
  float mx = red[0]; __syncthreads();
  float ssum = 0.f;
  for (int m = t; m < Mslots; m += 256) { float e = expf(sc[m] - mx); sc[m] = e; ssum += e; }
  red[t] = ssum; __syncthreads();
  for (int st = 128; st >= 1; st >>= 1) { if (t < st) red[t] += red[t + st]; __syncthreads(); }
  float inv = 1.f / red[0]; __syncthreads();
  float acc = 0.f;
  for (int m = 0; m < Mslots; ++m)
    acc += sc[m] * hidden[(long)(b * Tn + sidx[m]) * Hn + t];
  ctxv[b * Hn + t] = acc * inv;
}

// -------------------------------------------------- ctx @ Wout (WMMA, 1 pass)
__global__ __launch_bounds__(128) void outproj_kernel(
    const float* ctxv, const float* Wout, const float* bout, float* out) {
  const int lane = threadIdx.x & 31, wid = threadIdx.x >> 5;
  const int tile = blockIdx.x * (blockDim.x >> 5) + wid;
  const int mt = tile & 1, nt = tile >> 1;
  const int nlo = lane & 15, kh = lane >> 4;
  const int n0 = nt * 16 + nlo;
  v16bf a[8];
#pragma unroll
  for (int kt = 0; kt < 8; ++kt) {
    const float* row = ctxv + (mt * 16 + (lane & 15)) * Hn + kt * 32;
    v16bf f;
#pragma unroll
    for (int v = 0; v < 8; ++v) {
      int kv = (v < 4) ? (kh * 8 + 2 * v) : (16 + kh * 8 + 2 * (v - 4));
      f[2 * v] = tobf(row[kv]); f[2 * v + 1] = tobf(row[kv + 1]);
    }
    a[kt] = f;
  }
  float bias = bout[n0];
  v8f acc0, acc1;
#pragma unroll
  for (int v = 0; v < 8; ++v) { acc0[v] = bias; acc1[v] = 0.f; }
#pragma unroll
  for (int kt = 0; kt < 8; kt += 2) {
    v16bf bf0, bf1;
#pragma unroll
    for (int v = 0; v < 8; ++v) {
      int k0 = kt * 32 + ((v < 4) ? (kh * 8 + 2 * v) : (16 + kh * 8 + 2 * (v - 4)));
      int k1 = k0 + 32;
      bf0[2 * v]     = tobf(Wout[(long)k0 * Vn + n0]);
      bf0[2 * v + 1] = tobf(Wout[(long)(k0 + 1) * Vn + n0]);
      bf1[2 * v]     = tobf(Wout[(long)k1 * Vn + n0]);
      bf1[2 * v + 1] = tobf(Wout[(long)(k1 + 1) * Vn + n0]);
    }
    acc0 = wmma_bf16(a[kt], bf0, acc0);
    acc1 = wmma_bf16(a[kt + 1], bf1, acc1);
  }
#pragma unroll
  for (int v = 0; v < 8; ++v) {
    int m = mt * 16 + v + 8 * kh;
    out[(long)m * Vn + n0] = acc0[v] + acc1[v];
  }
}

extern "C" void kernel_launch(void* const* d_in, const int* in_sizes, int n_in,
                              void* d_out, int out_size, void* d_ws, size_t ws_size,
                              hipStream_t stream) {
  const int*   seq   = (const int*)d_in[0];
  const float* embed = (const float*)d_in[1];
  const float* W1    = (const float*)d_in[2];
  const float* b1    = (const float*)d_in[3];
  const float* W2    = (const float*)d_in[4];
  const float* b2    = (const float*)d_in[5];
  const float* ln_g  = (const float*)d_in[6];
  const float* ln_b  = (const float*)d_in[7];
  const float* Wg    = (const float*)d_in[8];
  const float* bg    = (const float*)d_in[9];
  const float* Wr1   = (const float*)d_in[10];
  const float* br1   = (const float*)d_in[11];
  const float* Wr2   = (const float*)d_in[12];
  const float* br2   = (const float*)d_in[13];
  const float* Wq    = (const float*)d_in[14];
  const float* bq    = (const float*)d_in[15];
  const float* Wout  = (const float*)d_in[16];
  const float* bout  = (const float*)d_in[17];
  float* out = (float*)d_out;

  char* ws = (char*)d_ws;
  size_t off = 0;
  auto carve = [&](size_t bytes) -> void* {
    void* p = ws + off;
    off += (bytes + 255) & ~(size_t)255;
    return p;
  };
  __bf16* w1p     = (__bf16*)carve((size_t)Hn * H2n * 2);
  __bf16* w2p     = (__bf16*)carve((size_t)Hn * H2n * 2);
  __bf16* wr1p    = (__bf16*)carve((size_t)Hn * H2n * 2);
  float*  hidden  = (float*)carve((size_t)Bn * Tn * Hn * 4);
  float*  fscore  = (float*)carve((size_t)Bn * Tn * 4);
  float*  rscores = (float*)carve((size_t)Bn * Tn * 4);
  float*  ctxsum  = (float*)carve((size_t)Bn * Hn * 4);
  float*  context = (float*)carve((size_t)Bn * Hn * 4);
  unsigned char* sel = (unsigned char*)carve((size_t)Bn * Tn);
  int*    slot_idx = (int*)carve((size_t)Bn * Mslots * 4);
  float*  ctxv    = (float*)carve((size_t)Bn * Hn * 4);

  pack_b<<<(8192 + 255) / 256, 256, 0, stream>>>(W1, w1p, Hn, H2n);    // K=256,N=512
  pack_b<<<(8192 + 255) / 256, 256, 0, stream>>>(W2, w2p, H2n, Hn);    // K=512,N=256
  pack_b<<<(8192 + 255) / 256, 256, 0, stream>>>(Wr1, wr1p, H2n, Hn);  // K=512,N=256
  hipMemsetAsync(ctxsum, 0, (size_t)Bn * Hn * 4, stream);
  hipMemsetAsync(sel, 0, (size_t)Bn * Tn, stream);

  const size_t lds_ffn = 4 * (16 * Hn * 4 + 16 * H2n * 2);   // 128 KB / block
  ffn_ln_kernel<<<(Bn * Tn / 16) / 4, 128, lds_ffn, stream>>>(
      seq, embed, b1, b2, ln_g, ln_b, Wg, bg, w1p, w2p, hidden, fscore, ctxsum);

  finalize_ctx<<<(Bn * Hn + 255) / 256, 256, 0, stream>>>(ctxsum, context);

  const size_t lds_gate = 4 * (16 * Hn * 4 + Hn * 4);        // 68 KB / block
  gate_kernel<<<(Bn * 256) / 4, 128, lds_gate, stream>>>(
      hidden, context, wr1p, br1, Wr2, br2, rscores);

  topk_kernel<<<Bn, 256, 0, stream>>>(fscore, nullptr, sel, slot_idx, 0, NCAND, FSn);
  topk_kernel<<<Bn, 256, 0, stream>>>(rscores, sel, nullptr, slot_idx, FSn, NCAND, RETRO);

  attn_kernel<<<Bn, 256, 0, stream>>>(hidden, slot_idx, Wq, bq, ctxv);

  outproj_kernel<<<(2 * (Vn / 16)) / 4, 128, 0, stream>>>(ctxv, Wout, bout, out);
}
